// Graph_MLP_10110353015255
// MI455X (gfx1250) — matrix operand order repaired and checked by reading, not compiled, not run on hardware
//
#include <hip/hip_runtime.h>

#define NB 2
#define NP 16384
#define NK 16

typedef _Float16 half_t;
typedef __attribute__((ext_vector_type(16))) _Float16 v16h;
typedef __attribute__((ext_vector_type(8)))  _Float16 v8h;
typedef __attribute__((ext_vector_type(8)))  float    v8f;

// ---------------------------------------------------------------------------
// Fast tanh: CDNA5 V_TANH_F32 (single TRANS op), branchless fallback.
// ---------------------------------------------------------------------------
__device__ __forceinline__ float fast_tanh(float x) {
#if __has_builtin(__builtin_amdgcn_tanhf)
    return __builtin_amdgcn_tanhf(x);
#elif __has_builtin(__builtin_amdgcn_tanh_f32)
    return __builtin_amdgcn_tanh_f32(x);
#else
    float e = __expf(2.0f * x);
    return 1.0f - 2.0f / (e + 1.0f);
#endif
}

// ---------------------------------------------------------------------------
// WMMA fragment helpers (layouts per CDNA5 ISA 7.12.2, wave32)
// ---------------------------------------------------------------------------
__device__ __forceinline__ v16h cat16(v8h a, v8h b) {
    v16h r;
    *(v8h*)&r       = a;
    *((v8h*)&r + 1) = b;
    return r;
}

// A: 16x32 f16, row-major source. lane(lo,hi): row = mbase+lo,
// K = {kbase+hi*8 .. +7} then {kbase+16+hi*8 .. +7}  (two 16B LDS loads)
__device__ __forceinline__ v16h load_a_frag(const half_t* s, int ld, int mbase,
                                            int kbase, int lo, int hi) {
    const half_t* p = s + (mbase + lo) * ld + kbase + hi * 8;
    return cat16(*(const v8h*)p, *(const v8h*)(p + 16));
}

// B: 32x16 f16. Weights stored transposed WT[out][in] so lane(lo,hi):
// col N = nbase+lo, K = {kbase+hi*8 .. +7} then {kbase+16+hi*8 .. +7} (two 16B LDS loads)
__device__ __forceinline__ v16h load_b_frag(const half_t* s, int ld, int nbase,
                                            int kbase, int lo, int hi) {
    const half_t* p = s + (nbase + lo) * ld + kbase + hi * 8;
    return cat16(*(const v8h*)p, *(const v8h*)(p + 16));
}

__device__ __forceinline__ v8f wmma16(v16h a, v16h b, v8f c) {
    return __builtin_amdgcn_wmma_f32_16x16x32_f16(false, a, false, b,
                                                  (short)0, c, false, false);
}

__device__ __forceinline__ void copy_lds(half_t* dst, const half_t* __restrict__ src,
                                         int nHalves, int tid, int nthr) {
    int nv = nHalves >> 3;
    for (int i = tid; i < nv; i += nthr)
        *(v8h*)(dst + (i << 3)) = *(const v8h*)(src + (i << 3));
}

// One hidden layer: per-wave 16-row strip, all NT column tiles.
// D-layout (32b 16x16): VGPR r, lane(lo,hi) -> (m = mbase+hi*8+r, n = nbase+lo)
template<int KT, int NT, bool TANH>
__device__ __forceinline__ void mlp_layer(const half_t* sIn, int ldIn,
                                          const half_t* sWt,
                                          const float* __restrict__ bias,
                                          half_t* sOut, int ldOut,
                                          int mbase, int lo, int hi) {
    float bb[NT];                         // batch bias loads up front
    #pragma unroll
    for (int nt = 0; nt < NT; ++nt) bb[nt] = bias[nt * 16 + lo];
    v16h aF[KT];
    #pragma unroll
    for (int kt = 0; kt < KT; ++kt)
        aF[kt] = load_a_frag(sIn, ldIn, mbase, kt * 32, lo, hi);
    #pragma unroll
    for (int nt = 0; nt < NT; ++nt) {
        v8f acc = {0.f,0.f,0.f,0.f,0.f,0.f,0.f,0.f};
        #pragma unroll
        for (int kt = 0; kt < KT; ++kt)
            acc = wmma16(aF[kt], load_b_frag(sWt, KT * 32, nt * 16, kt * 32, lo, hi), acc);
        #pragma unroll
        for (int r = 0; r < 8; ++r) {
            float v = acc[r] + bb[nt];
            if (TANH) v = fast_tanh(v);
            sOut[(mbase + hi * 8 + r) * ldOut + nt * 16 + lo] = (half_t)v;
        }
    }
}

// ---------------------------------------------------------------------------
// Prep kernels
// ---------------------------------------------------------------------------
__global__ void wtpad_kernel(const float* __restrict__ W, half_t* __restrict__ WT,
                             int in_dim, int out_dim, int inPad, int outPad) {
    int idx = blockIdx.x * 256 + threadIdx.x;
    if (idx >= inPad * outPad) return;
    int o = idx / inPad, i = idx - o * inPad;
    float v = (i < in_dim && o < out_dim) ? W[i * out_dim + o] : 0.0f;
    WT[idx] = (half_t)v;
}

__global__ void build_point_f_kernel(const float* __restrict__ dyn,
                                     const float* __restrict__ ref,
                                     const float* __restrict__ stiff,
                                     const float* __restrict__ mass,
                                     float* __restrict__ pf) {
    int i = blockIdx.x * 256 + threadIdx.x;
    if (i >= NB * NP) return;
    const float* d  = dyn + (size_t)i * 9;
    const float* rf = ref + (size_t)i * 9;
    float rc0 = rf[3], rc1 = rf[4], rc2 = rf[5];
    float* o = pf + (size_t)i * 18;
    o[0]  = d[0] - rc0;  o[1]  = d[1] - rc1;  o[2]  = d[2] - rc2;
    o[3]  = d[3] - rc0;  o[4]  = d[4] - rc1;  o[5]  = d[5] - rc2;
    o[6]  = d[6] - rc0;  o[7]  = d[7] - rc1;  o[8]  = d[8] - rc2;
    o[9]  = rf[0] - rc0; o[10] = rf[1] - rc1; o[11] = rf[2] - rc2;
    o[12] = rf[6] - rc0; o[13] = rf[7] - rc1; o[14] = rf[8] - rc2;
    float s = stiff[i], m = mass[i];
    o[15] = s; o[16] = m; o[17] = s / m;
}

// ---------------------------------------------------------------------------
// Edge MLP: 524288 rows x [37->128->128->128->128->128], masked K-sum.
// Per-layer weight staging: LDS = 32KB W + 2x32KB act = 96KB -> 3 WGs/WGP
// (6 waves/SIMD latency hiding; same total L2 weight traffic as preloading).
// ---------------------------------------------------------------------------
__global__ void __launch_bounds__(256)
edge_mlp_kernel(const int* __restrict__ constraint,
                const float* __restrict__ dyn,
                const float* __restrict__ ref,
                const int* __restrict__ adj,
                const float* __restrict__ point_f,
                const half_t* __restrict__ wAll,
                const float* __restrict__ eb0, const float* __restrict__ eb1,
                const float* __restrict__ eb2, const float* __restrict__ eb3,
                const float* __restrict__ eb4,
                float* __restrict__ out_edge) {
    extern __shared__ __align__(16) char smem[];
    half_t* sW   = (half_t*)smem;        // 16384 halves (one layer, transposed)
    half_t* sA0  = sW + 16384;           // 128x128 f16
    half_t* sA1  = sA0 + 16384;          // 128x128 f16
    float*  sRed = (float*)sA0;          // 128x128 f32 (aliases sA0+sA1)
    __shared__ float sMask[128];

    const int tid = threadIdx.x;
    copy_lds(sW, wAll, 8192, tid, 256);  // layer 0 weights (64x128)

    if (tid < 128) {                     // build edge feature row (37 + pad to 64)
        int e   = blockIdx.x * 128 + tid;
        int b   = e / (NP * NK);
        int rem = e - b * (NP * NK);
        int n   = rem >> 4;
        int k   = rem & 15;
        int j   = adj[n * NK + k];
        const float* rcp = ref + ((size_t)(b * NP + n)) * 9 + 3;
        float rc3[3] = {rcp[0], rcp[1], rcp[2]};
        const float* dn = dyn + ((size_t)(b * NP + j)) * 9;
        const float* rn = ref + ((size_t)(b * NP + j)) * 9;
        half_t* row = sA0 + tid * 64;
        row[0] = (half_t)(float)constraint[j];
        #pragma unroll
        for (int q = 0; q < 9; ++q) row[1 + q]  = (half_t)(dn[q] - rc3[q % 3]);
        #pragma unroll
        for (int q = 0; q < 9; ++q) row[10 + q] = (half_t)(rn[q] - rc3[q % 3]);
        const float* pf = point_f + ((size_t)(b * NP + n)) * 18;
        #pragma unroll
        for (int q = 0; q < 18; ++q) row[19 + q] = (half_t)pf[q];
        #pragma unroll
        for (int q = 37; q < 64; ++q) row[q] = (half_t)0.0f;
        sMask[tid] = (j != 0) ? 1.0f : 0.0f;
    }
    __syncthreads();

    const int lane  = tid & 31;
    const int mbase = (tid >> 5) * 16;
    const int lo = lane & 15, hi = lane >> 4;

    mlp_layer<2, 8, true>(sA0,  64, sW, eb0, sA1, 128, mbase, lo, hi);
    __syncthreads();
    copy_lds(sW, wAll + 8192, 16384, tid, 256);
    __syncthreads();
    mlp_layer<4, 8, true>(sA1, 128, sW, eb1, sA0, 128, mbase, lo, hi);
    __syncthreads();
    copy_lds(sW, wAll + 24576, 16384, tid, 256);
    __syncthreads();
    mlp_layer<4, 8, true>(sA0, 128, sW, eb2, sA1, 128, mbase, lo, hi);
    __syncthreads();
    copy_lds(sW, wAll + 40960, 16384, tid, 256);
    __syncthreads();
    mlp_layer<4, 8, true>(sA1, 128, sW, eb3, sA0, 128, mbase, lo, hi);
    __syncthreads();
    copy_lds(sW, wAll + 57344, 16384, tid, 256);
    __syncthreads();

    // final layer (linear): keep accumulators in registers, then masked fp32 dump
    float bb4[8];
    #pragma unroll
    for (int nt = 0; nt < 8; ++nt) bb4[nt] = eb4[nt * 16 + lo];
    v16h aF[4];
    #pragma unroll
    for (int kt = 0; kt < 4; ++kt)
        aF[kt] = load_a_frag(sA0, 128, mbase, kt * 32, lo, hi);
    v8f accs[8];
    #pragma unroll
    for (int nt = 0; nt < 8; ++nt) {
        v8f acc = {0.f,0.f,0.f,0.f,0.f,0.f,0.f,0.f};
        #pragma unroll
        for (int kt = 0; kt < 4; ++kt)
            acc = wmma16(aF[kt], load_b_frag(sW, 128, nt * 16, kt * 32, lo, hi), acc);
        accs[nt] = acc;
    }
    __syncthreads();                     // all A-reads done before sRed overwrite
    #pragma unroll
    for (int nt = 0; nt < 8; ++nt) {
        #pragma unroll
        for (int r = 0; r < 8; ++r) {
            int m = mbase + hi * 8 + r;
            sRed[m * 128 + nt * 16 + lo] = (accs[nt][r] + bb4[nt]) * sMask[m];
        }
    }
    __syncthreads();
    size_t gp0 = (size_t)blockIdx.x * 8;  // 8 points per tile
    for (int i = tid; i < 1024; i += 256) {
        int p = i >> 7, c = i & 127;
        float s = 0.0f;
        #pragma unroll
        for (int r = 0; r < 16; ++r) s += sRed[(p * 16 + r) * 128 + c];
        out_edge[(gp0 + p) * 128 + c] = s;
    }
}

// ---------------------------------------------------------------------------
// Point MLP: 32768 rows x [18->64->64->64->64->64]
// Per-layer staging: LDS = 8KB W + 2x16KB act = 40KB
// ---------------------------------------------------------------------------
__global__ void __launch_bounds__(256)
point_mlp_kernel(const float* __restrict__ point_f,
                 const half_t* __restrict__ wAll,
                 const float* __restrict__ pb0, const float* __restrict__ pb1,
                 const float* __restrict__ pb2, const float* __restrict__ pb3,
                 const float* __restrict__ pb4,
                 float* __restrict__ out_point) {
    extern __shared__ __align__(16) char smem[];
    half_t* sW  = (half_t*)smem;         // 4096 halves (one layer)
    half_t* sA0 = sW + 4096;             // 128x64
    half_t* sA1 = sA0 + 8192;            // 128x64

    const int tid = threadIdx.x;
    copy_lds(sW, wAll, 2048, tid, 256);  // layer 0 weights (32x64)

    if (tid < 128) {
        size_t row = (size_t)blockIdx.x * 128 + tid;
        const float* pf = point_f + row * 18;
        half_t* rp = sA0 + tid * 32;
        #pragma unroll
        for (int q = 0; q < 18; ++q) rp[q] = (half_t)pf[q];
        #pragma unroll
        for (int q = 18; q < 32; ++q) rp[q] = (half_t)0.0f;
    }
    __syncthreads();

    const int lane  = tid & 31;
    const int mbase = (tid >> 5) * 16;
    const int lo = lane & 15, hi = lane >> 4;

    mlp_layer<1, 4, true>(sA0, 32, sW, pb0, sA1, 64, mbase, lo, hi);
    __syncthreads();
    copy_lds(sW, wAll + 2048, 4096, tid, 256);
    __syncthreads();
    mlp_layer<2, 4, true>(sA1, 64, sW, pb1, sA0, 64, mbase, lo, hi);
    __syncthreads();
    copy_lds(sW, wAll + 6144, 4096, tid, 256);
    __syncthreads();
    mlp_layer<2, 4, true>(sA0, 64, sW, pb2, sA1, 64, mbase, lo, hi);
    __syncthreads();
    copy_lds(sW, wAll + 10240, 4096, tid, 256);
    __syncthreads();
    mlp_layer<2, 4, true>(sA1, 64, sW, pb3, sA0, 64, mbase, lo, hi);
    __syncthreads();
    copy_lds(sW, wAll + 14336, 4096, tid, 256);
    __syncthreads();

    // final layer (linear) -> global fp32
    float bb4[4];
    #pragma unroll
    for (int nt = 0; nt < 4; ++nt) bb4[nt] = pb4[nt * 16 + lo];
    v16h aF[2];
    #pragma unroll
    for (int kt = 0; kt < 2; ++kt)
        aF[kt] = load_a_frag(sA0, 64, mbase, kt * 32, lo, hi);
    size_t rowBase = (size_t)blockIdx.x * 128;
    #pragma unroll
    for (int nt = 0; nt < 4; ++nt) {
        v8f acc = {0.f,0.f,0.f,0.f,0.f,0.f,0.f,0.f};
        #pragma unroll
        for (int kt = 0; kt < 2; ++kt)
            acc = wmma16(aF[kt], load_b_frag(sW, 64, nt * 16, kt * 32, lo, hi), acc);
        #pragma unroll
        for (int r = 0; r < 8; ++r)
            out_point[(rowBase + mbase + hi * 8 + r) * 64 + nt * 16 + lo] = acc[r] + bb4[nt];
    }
}

// ---------------------------------------------------------------------------
// Instance MLP: 32768 rows x [192->192->192->192->192->3]
// Weights staged per layer (192x192 f16 = 72KB buffer)
// ---------------------------------------------------------------------------
__global__ void __launch_bounds__(256)
inst_mlp_kernel(const float* __restrict__ out_point,
                const float* __restrict__ out_edge,
                const half_t* __restrict__ wAll,
                const float* __restrict__ ib0, const float* __restrict__ ib1,
                const float* __restrict__ ib2, const float* __restrict__ ib3,
                const float* __restrict__ ib4,
                float* __restrict__ out) {
    extern __shared__ __align__(16) char smem[];
    half_t* sW  = (half_t*)smem;         // 36864 halves
    half_t* sA0 = sW + 36864;            // 128x192
    half_t* sA1 = sA0 + 24576;           // 128x192

    const int tid = threadIdx.x;
    if (tid < 128) {                     // concat(out_point, out_edge) -> f16
        size_t row = (size_t)blockIdx.x * 128 + tid;
        half_t* rp = sA0 + tid * 192;
        const float* op = out_point + row * 64;
        const float* oe = out_edge + row * 128;
        #pragma unroll 8
        for (int q = 0; q < 64; ++q) rp[q] = (half_t)op[q];
        #pragma unroll 8
        for (int q = 0; q < 128; ++q) rp[64 + q] = (half_t)oe[q];
    }
    copy_lds(sW, wAll, 36864, tid, 256);
    __syncthreads();

    const int lane  = tid & 31;
    const int mbase = (tid >> 5) * 16;
    const int lo = lane & 15, hi = lane >> 4;

    mlp_layer<6, 12, true>(sA0, 192, sW, ib0, sA1, 192, mbase, lo, hi);
    __syncthreads();
    copy_lds(sW, wAll + 36864, 36864, tid, 256);
    __syncthreads();
    mlp_layer<6, 12, true>(sA1, 192, sW, ib1, sA0, 192, mbase, lo, hi);
    __syncthreads();
    copy_lds(sW, wAll + 73728, 36864, tid, 256);
    __syncthreads();
    mlp_layer<6, 12, true>(sA0, 192, sW, ib2, sA1, 192, mbase, lo, hi);
    __syncthreads();
    copy_lds(sW, wAll + 110592, 36864, tid, 256);
    __syncthreads();
    mlp_layer<6, 12, true>(sA1, 192, sW, ib3, sA0, 192, mbase, lo, hi);
    __syncthreads();
    copy_lds(sW, wAll + 147456, 3072, tid, 256);   // 192x3 padded to 192x16
    __syncthreads();

    // final layer: single 16-col tile, keep cols 0..2
    v16h aF[6];
    #pragma unroll
    for (int kt = 0; kt < 6; ++kt)
        aF[kt] = load_a_frag(sA0, 192, mbase, kt * 32, lo, hi);
    v8f acc = {0.f,0.f,0.f,0.f,0.f,0.f,0.f,0.f};
    #pragma unroll
    for (int kt = 0; kt < 6; ++kt)
        acc = wmma16(aF[kt], load_b_frag(sW, 192, 0, kt * 32, lo, hi), acc);
    if (lo < 3) {
        float bb = ib4[lo];
        size_t rowBase = (size_t)blockIdx.x * 128;
        #pragma unroll
        for (int r = 0; r < 8; ++r)
            out[(rowBase + mbase + hi * 8 + r) * 3 + lo] = acc[r] + bb;
    }
}

// ---------------------------------------------------------------------------
// Host launcher
// ---------------------------------------------------------------------------
static inline void launch_wtpad(const void* W, half_t* WT, int in_d, int out_d,
                                int inPad, int outPad, hipStream_t s) {
    int total = inPad * outPad;
    wtpad_kernel<<<(total + 255) / 256, 256, 0, s>>>((const float*)W, WT,
                                                     in_d, out_d, inPad, outPad);
}

extern "C" void kernel_launch(void* const* d_in, const int* in_sizes, int n_in,
                              void* d_out, int out_size, void* d_ws, size_t ws_size,
                              hipStream_t stream) {
    const int*   constraint = (const int*)d_in[0];
    const float* dyn   = (const float*)d_in[1];
    const float* ref   = (const float*)d_in[2];
    const int*   adj   = (const int*)d_in[3];
    const float* stiff = (const float*)d_in[4];
    const float* mass  = (const float*)d_in[5];

    // locate the three MLPs among the flattened params by their unique W1 sizes
    int eb = -1, pb = -1, ib = -1;
    for (int i = 6; i < n_in; ++i) {
        if (in_sizes[i] == 37 * 128   && eb < 0) eb = i;
        if (in_sizes[i] == 18 * 64    && pb < 0) pb = i;
        if (in_sizes[i] == 192 * 192  && ib < 0) ib = i;
    }
    if (eb < 0 || pb < 0 || ib < 0) return;

    // workspace layout: f16 transposed weights, then fp32 activations
    half_t* wsH = (half_t*)d_ws;
    float*  wsF = (float*)((char*)d_ws + 491520);
    float* point_f   = wsF;                         // 2*16384*18
    float* out_point = wsF + 589824;                // 2*16384*64
    float* out_edge  = wsF + 589824 + 2097152;      // 2*16384*128

    // edge weights (in padded 37->64 on layer 0)
    launch_wtpad(d_in[eb + 0], wsH + 0,      37, 128,  64, 128, stream);
    launch_wtpad(d_in[eb + 2], wsH + 8192,  128, 128, 128, 128, stream);
    launch_wtpad(d_in[eb + 4], wsH + 24576, 128, 128, 128, 128, stream);
    launch_wtpad(d_in[eb + 6], wsH + 40960, 128, 128, 128, 128, stream);
    launch_wtpad(d_in[eb + 8], wsH + 57344, 128, 128, 128, 128, stream);
    // point weights (18->32 pad on layer 0)
    launch_wtpad(d_in[pb + 0], wsH + 73728,  18, 64, 32, 64, stream);
    launch_wtpad(d_in[pb + 2], wsH + 75776,  64, 64, 64, 64, stream);
    launch_wtpad(d_in[pb + 4], wsH + 79872,  64, 64, 64, 64, stream);
    launch_wtpad(d_in[pb + 6], wsH + 83968,  64, 64, 64, 64, stream);
    launch_wtpad(d_in[pb + 8], wsH + 88064,  64, 64, 64, 64, stream);
    // instance weights (out 3->16 pad on layer 4)
    launch_wtpad(d_in[ib + 0], wsH + 92160,  192, 192, 192, 192, stream);
    launch_wtpad(d_in[ib + 2], wsH + 129024, 192, 192, 192, 192, stream);
    launch_wtpad(d_in[ib + 4], wsH + 165888, 192, 192, 192, 192, stream);
    launch_wtpad(d_in[ib + 6], wsH + 202752, 192, 192, 192, 192, stream);
    launch_wtpad(d_in[ib + 8], wsH + 239616, 192,   3, 192,  16, stream);

    build_point_f_kernel<<<(NB * NP + 255) / 256, 256, 0, stream>>>(
        dyn, ref, stiff, mass, point_f);

    edge_mlp_kernel<<<(NB * NP * NK) / 128, 256, 98304, stream>>>(
        constraint, dyn, ref, adj, point_f, wsH + 0,
        (const float*)d_in[eb + 1], (const float*)d_in[eb + 3],
        (const float*)d_in[eb + 5], (const float*)d_in[eb + 7],
        (const float*)d_in[eb + 9], out_edge);

    point_mlp_kernel<<<(NB * NP) / 128, 256, 40960, stream>>>(
        point_f, wsH + 73728,
        (const float*)d_in[pb + 1], (const float*)d_in[pb + 3],
        (const float*)d_in[pb + 5], (const float*)d_in[pb + 7],
        (const float*)d_in[pb + 9], out_point);

    inst_mlp_kernel<<<(NB * NP) / 128, 256, 172032, stream>>>(
        out_point, out_edge, wsH + 92160,
        (const float*)d_in[ib + 1], (const float*)d_in[ib + 3],
        (const float*)d_in[ib + 5], (const float*)d_in[ib + 7],
        (const float*)d_in[ib + 9], (float*)d_out);

    (void)out_size; (void)ws_size;
}
